// NewFullBandGATWithNodeImportance_8899172237596
// MI455X (gfx1250) — matrix operand order repaired and checked by reading, not compiled, not run on hardware
//
#include <hip/hip_runtime.h>
#include <hip/hip_bf16.h>
#include <math.h>

// ---------------------------------------------------------------------------
// Types for CDNA5 WMMA (wave32, v_wmma_f32_16x16x32_bf16)
// ---------------------------------------------------------------------------
typedef __attribute__((ext_vector_type(16))) __bf16 v16bf;
typedef __attribute__((ext_vector_type(8)))  float  v8f;
typedef __attribute__((ext_vector_type(4)))  unsigned int u32x4;

__device__ __forceinline__ unsigned short f2bf(float f) {
  unsigned int x = __float_as_uint(f);
  unsigned int r = (x + 0x7FFFu + ((x >> 16) & 1u)) >> 16;   // RNE
  return (unsigned short)r;
}
__device__ __forceinline__ unsigned int pkbf(float a, float b) {
  return (unsigned int)f2bf(a) | ((unsigned int)f2bf(b) << 16);
}

union FragU { v16bf v; u32x4 q[2]; };

// Tile geometry: block = 256 threads = 8 waves; 64(M) x 128(N) tile, K-step 32.
// Each wave (wm 0..1, wn 0..3) owns a 32x32 output tile (2x2 WMMA frags).
#define TM 64
#define TN 128
#define TK 32
#define TKP 40   // padded LDS row stride (80B): 20*row mod 64 banks -> conflict-free b128 reads

// ---------------------------------------------------------------------------
// Generic tiled GEMM: C[M,N] = act(A[M,K] @ B[K,N] + bias)
// Requires M%64==0, N%128==0, K%32==0 (true for all uses here).
// ---------------------------------------------------------------------------
__global__ __launch_bounds__(256) void gemm_bf16_wmma(
    const float* __restrict__ A, const float* __restrict__ B,
    const float* __restrict__ bias, float* __restrict__ C,
    int M, int N, int K, int act)
{
  __shared__ __align__(16) unsigned short Al[TM][TKP];
  __shared__ __align__(16) unsigned short Bl[TN][TKP];   // transposed [n][k]
  const int tid  = threadIdx.x;
  const int lane = tid & 31;
  const int wave = tid >> 5;
  const int wm   = wave >> 2;      // 0..1
  const int wn   = wave & 3;       // 0..3
  const int blockM = blockIdx.y * TM;
  const int blockN = blockIdx.x * TN;
  const int half = lane >> 4;      // 0|1 selects K sub-chunk per ISA layout
  const int l16  = lane & 15;

  v8f acc[2][2] = {};

  for (int k0 = 0; k0 < K; k0 += TK) {
    // A: 64 rows x 32 k, K-contiguous -> 8-elem chunks: 2x float4 load, 1x b128 store
    for (int i = tid; i < (TM * TK) / 8; i += 256) {
      int r = i >> 2, c = (i & 3) * 8;
      const float4* ap = (const float4*)&A[(size_t)(blockM + r) * K + k0 + c];
      float4 v0 = ap[0], v1 = ap[1];
      u32x4 pk = { pkbf(v0.x, v0.y), pkbf(v0.z, v0.w), pkbf(v1.x, v1.y), pkbf(v1.z, v1.w) };
      *(u32x4*)&Al[r][c] = pk;
    }
    // B: 32 k-rows x 128 n, N-contiguous -> float4 along n, scatter 4 b16 into [n][k]
    for (int i = tid; i < (TK * TN) / 4; i += 256) {
      int kk = i >> 5, n4 = (i & 31) * 4;
      float4 v = *(const float4*)&B[(size_t)(k0 + kk) * N + blockN + n4];
      Bl[n4 + 0][kk] = f2bf(v.x);
      Bl[n4 + 1][kk] = f2bf(v.y);
      Bl[n4 + 2][kk] = f2bf(v.z);
      Bl[n4 + 3][kk] = f2bf(v.w);
    }
    __syncthreads();

    v16bf afrag[2], bfrag[2];
    #pragma unroll
    for (int i = 0; i < 2; ++i) {
      // A 16x32 bf16 layout: lanes 0-15 M=row K={0..7,16..23}; lanes 16-31 K={8..15,24..31}
      int row = wm * 32 + i * 16 + l16;
      const u32x4* p = (const u32x4*)&Al[row][half * 8];
      FragU u; u.q[0] = p[0]; u.q[1] = p[2];
      afrag[i] = u.v;
    }
    #pragma unroll
    for (int j = 0; j < 2; ++j) {
      // B 32x16 bf16 layout: N = lane%16, lanes 0-15 K={0..7,16..23}; lanes 16-31 K={8..15,24..31}
      int col = wn * 32 + j * 16 + l16;
      const u32x4* p = (const u32x4*)&Bl[col][half * 8];
      FragU u; u.q[0] = p[0]; u.q[1] = p[2];
      bfrag[j] = u.v;
    }
    #pragma unroll
    for (int i = 0; i < 2; ++i)
      #pragma unroll
      for (int j = 0; j < 2; ++j)
        acc[i][j] = __builtin_amdgcn_wmma_f32_16x16x32_bf16(
            false, afrag[i], false, bfrag[j], (short)0, acc[i][j], false, false);
    __syncthreads();
  }

  // C/D layout: VGPR r -> M = r (+8 for lanes 16-31), N = lane%16
  #pragma unroll
  for (int i = 0; i < 2; ++i)
    #pragma unroll
    for (int j = 0; j < 2; ++j) {
      int baseM = blockM + wm * 32 + i * 16 + (half ? 8 : 0);
      int baseN = blockN + wn * 32 + j * 16 + l16;
      float bv = bias ? bias[baseN] : 0.f;
      #pragma unroll
      for (int r = 0; r < 8; ++r) {
        float o = acc[i][j][r] + bv;
        if (act) o = fmaxf(o, 0.f);
        C[(size_t)(baseM + r) * N + baseN] = o;
      }
    }
}

// ---------------------------------------------------------------------------
// GAE adjacency: adj[g,i,j] = sigmoid( dot(r[g,i,:], r[g,j,:]) ), r = [32,512,128]
// B = A^T, so both LDS tiles are K-contiguous row copies of r (packed b128 staging).
// ---------------------------------------------------------------------------
__global__ __launch_bounds__(256) void adj_sigmoid_wmma(
    const float* __restrict__ r, float* __restrict__ adj)
{
  __shared__ __align__(16) unsigned short Al[TM][TKP];
  __shared__ __align__(16) unsigned short Bl[TN][TKP];
  const int tid  = threadIdx.x;
  const int lane = tid & 31;
  const int wave = tid >> 5;
  const int wm   = wave >> 2;
  const int wn   = wave & 3;
  const int g = blockIdx.z;
  const int blockM = blockIdx.y * TM;
  const int blockN = blockIdx.x * TN;
  const int half = lane >> 4;
  const int l16  = lane & 15;
  const float* rg = r + (size_t)g * 512 * 128;

  v8f acc[2][2] = {};

  for (int k0 = 0; k0 < 128; k0 += TK) {
    for (int i = tid; i < (TM * TK) / 8; i += 256) {
      int rr = i >> 2, c = (i & 3) * 8;
      const float4* ap = (const float4*)&rg[(size_t)(blockM + rr) * 128 + k0 + c];
      float4 v0 = ap[0], v1 = ap[1];
      u32x4 pk = { pkbf(v0.x, v0.y), pkbf(v0.z, v0.w), pkbf(v1.x, v1.y), pkbf(v1.z, v1.w) };
      *(u32x4*)&Al[rr][c] = pk;
    }
    for (int i = tid; i < (TN * TK) / 8; i += 256) {
      int rr = i >> 2, c = (i & 3) * 8;
      const float4* bp = (const float4*)&rg[(size_t)(blockN + rr) * 128 + k0 + c];
      float4 v0 = bp[0], v1 = bp[1];
      u32x4 pk = { pkbf(v0.x, v0.y), pkbf(v0.z, v0.w), pkbf(v1.x, v1.y), pkbf(v1.z, v1.w) };
      *(u32x4*)&Bl[rr][c] = pk;
    }
    __syncthreads();

    v16bf afrag[2], bfrag[2];
    #pragma unroll
    for (int i = 0; i < 2; ++i) {
      int row = wm * 32 + i * 16 + l16;
      const u32x4* p = (const u32x4*)&Al[row][half * 8];
      FragU u; u.q[0] = p[0]; u.q[1] = p[2];
      afrag[i] = u.v;
    }
    #pragma unroll
    for (int j = 0; j < 2; ++j) {
      int col = wn * 32 + j * 16 + l16;
      const u32x4* p = (const u32x4*)&Bl[col][half * 8];
      FragU u; u.q[0] = p[0]; u.q[1] = p[2];
      bfrag[j] = u.v;
    }
    #pragma unroll
    for (int i = 0; i < 2; ++i)
      #pragma unroll
      for (int j = 0; j < 2; ++j)
        acc[i][j] = __builtin_amdgcn_wmma_f32_16x16x32_bf16(
            false, afrag[i], false, bfrag[j], (short)0, acc[i][j], false, false);
    __syncthreads();
  }

  float* out = adj + (size_t)g * 512 * 512;
  #pragma unroll
  for (int i = 0; i < 2; ++i)
    #pragma unroll
    for (int j = 0; j < 2; ++j) {
      int baseM = blockM + wm * 32 + i * 16 + (half ? 8 : 0);
      int baseN = blockN + wn * 32 + j * 16 + l16;
      #pragma unroll
      for (int rr = 0; rr < 8; ++rr) {
        float v = acc[i][j][rr];
        out[(size_t)(baseM + rr) * 512 + baseN] = 1.f / (1.f + expf(-v));
      }
    }
}

// ---------------------------------------------------------------------------
// Scalar helper kernels
// ---------------------------------------------------------------------------
__global__ void fill_kernel(float* p, float v, int n) {
  int i = blockIdx.x * 256 + threadIdx.x;
  if (i < n) p[i] = v;
}

__device__ __forceinline__ void atomicMaxF(float* addr, float val) {
  if (val >= 0.f) atomicMax((int*)addr, __float_as_int(val));
  else            atomicMin((unsigned int*)addr, __float_as_uint(val));
}

// per-node attention scores: ss = h . a_s, sd = h . a_d  (one wave per node)
__global__ void node_scores(const float* __restrict__ h,
                            const float* __restrict__ as_,
                            const float* __restrict__ ad_,
                            float* __restrict__ ss, float* __restrict__ sd, int C)
{
  int node = blockIdx.x * 8 + (threadIdx.x >> 5);
  int lane = threadIdx.x & 31;
  const float* hp = h + (size_t)node * C;
  float a = 0.f, b = 0.f;
  for (int c = lane; c < C; c += 32) { float hv = hp[c]; a += hv * as_[c]; b += hv * ad_[c]; }
  for (int off = 16; off; off >>= 1) { a += __shfl_xor(a, off); b += __shfl_xor(b, off); }
  if (lane == 0) { ss[node] = a; sd[node] = b; }
}

__global__ void edge_logits(const int* __restrict__ ei,
                            const float* __restrict__ ss, const float* __restrict__ sd,
                            float* __restrict__ e, float* __restrict__ mbuf,
                            int E, int Nn)
{
  int i = blockIdx.x * 256 + threadIdx.x;
  if (i >= E + Nn) return;
  int s, d;
  if (i < E) { s = ei[i]; d = ei[E + i]; } else { s = d = i - E; }
  float ev = ss[s] + sd[d];
  ev = ev > 0.f ? ev : 0.2f * ev;            // leaky_relu slope 0.2
  e[i] = ev;
  atomicMaxF(&mbuf[d], ev);
}

__global__ void edge_exp(const int* __restrict__ ei,
                         const float* __restrict__ e, const float* __restrict__ mbuf,
                         float* __restrict__ ex, float* __restrict__ denom,
                         int E, int Nn)
{
  int i = blockIdx.x * 256 + threadIdx.x;
  if (i >= E + Nn) return;
  int d = (i < E) ? ei[E + i] : (i - E);
  float xv = expf(e[i] - mbuf[d]);
  ex[i] = xv;
  atomicAdd(&denom[d], xv);
}

// aggregate: agg[dst] += h[src] * alpha  (one wave per edge, L2-resident atomics)
__global__ void edge_aggregate(const int* __restrict__ ei,
                               const float* __restrict__ h,
                               const float* __restrict__ ex, const float* __restrict__ denom,
                               float* __restrict__ agg, int E, int Nn, int C)
{
  int edge = blockIdx.x * 8 + (threadIdx.x >> 5);
  int lane = threadIdx.x & 31;
  if (edge >= E + Nn) return;
  int s, d;
  if (edge < E) { s = ei[edge]; d = ei[E + edge]; } else { s = d = edge - E; }
  float alpha = ex[edge] / (denom[d] + 1e-16f);
  const float* hs = h + (size_t)s * C;
  float* od = agg + (size_t)d * C;
  for (int c = lane; c < C; c += 32) atomicAdd(&od[c], hs[c] * alpha);
}

// z = relu(((agg + b) / sqrt(1+eps)) * g + beta), in place
__global__ void bias_bn_relu(float* __restrict__ z, const float* __restrict__ b,
                             const float* __restrict__ g, const float* __restrict__ beta,
                             int C, int total)
{
  int i = blockIdx.x * 256 + threadIdx.x;
  if (i >= total) return;
  int c = i & (C - 1);                       // C is a power of two
  const float inv = 0.99999500003749969f;    // 1/sqrt(1+1e-5)
  float v = (z[i] + b[c]) * inv * g[c] + beta[c];
  z[i] = fmaxf(v, 0.f);
}

// global mean pool over contiguous 512-node graphs, H=128
__global__ void pool_mean(const float* __restrict__ z, float* __restrict__ pooled) {
  int g = blockIdx.x, c = threadIdx.x;       // block 128
  float s = 0.f;
  const float* zp = z + (size_t)g * 512 * 128 + c;
  for (int n = 0; n < 512; ++n) s += zp[(size_t)n * 128];
  pooled[g * 128 + c] = s * (1.f / 512.f);
}

// tiny head GEMM with optional row-concat input: C = act(concat(A1,A2) @ B + bias)
__global__ void head_gemm(const float* __restrict__ A1, int K1,
                          const float* __restrict__ A2, int K2,
                          const float* __restrict__ B, const float* __restrict__ bias,
                          float* __restrict__ C, int M, int N, int act)
{
  int idx = blockIdx.x * 256 + threadIdx.x;
  if (idx >= M * N) return;
  int m = idx / N, n = idx % N;
  float s = bias ? bias[n] : 0.f;
  for (int k = 0; k < K1; ++k) s += A1[m * K1 + k] * B[k * N + n];
  for (int k = 0; k < K2; ++k) s += A2[m * K2 + k] * B[(K1 + k) * N + n];
  if (act) s = fmaxf(s, 0.f);
  C[idx] = s;
}

// ---------------------------------------------------------------------------
// Host orchestration
// ---------------------------------------------------------------------------
extern "C" void kernel_launch(void* const* d_in, const int* in_sizes, int n_in,
                              void* d_out, int out_size, void* d_ws, size_t ws_size,
                              hipStream_t stream)
{
  (void)in_sizes; (void)n_in; (void)out_size; (void)ws_size;
  const int NN = 16384, EE = 262144, ET = EE + NN, GG = 32;

  const float* x  = (const float*)d_in[0];
  const int*   ei = (const int*)d_in[1];
  const float* gf = (const float*)d_in[3];
  auto P = [&](int i) { return (const float*)d_in[i]; };
  // params: 4 + 6*l : gat{l}_W, as, ad, b, bn_g, bn_b ; then 22:Wq 23:Wk 24:Wv
  // 25:mlp_W 26:mlp_b 27..34: c1..c4 W/b ; 35:g1_W 36:g1_b 37:g2_W 38:g2_b

  // workspace layout (floats)
  float* ws = (float*)d_ws;
  size_t off = 0;
  auto alloc = [&](size_t n) { float* p = ws + off; off += n; return p; };
  float* Z0    = alloc((size_t)NN * 512);
  float* Z1    = alloc((size_t)NN * 512);
  float* HB    = alloc((size_t)NN * 512);
  float* ss    = alloc(NN);
  float* sd    = alloc(NN);
  float* mbuf  = alloc(NN);
  float* dbuf  = alloc(NN);
  float* ebuf  = alloc(ET);
  float* exbuf = alloc(ET);
  float* pooled= alloc(GG * 128);
  float* vbuf  = alloc(GG * 128);
  float* fused = alloc(GG * 128);
  float* h1    = alloc(GG * 64);
  float* h2    = alloc(GG * 16);
  float* h3    = alloc(GG * 8);

  const int Cin[3]  = {64, 512, 256};
  const int Cout[3] = {512, 256, 128};
  const float* zin[3]  = {x, Z0, Z1};
  float*       zout[3] = {Z0, Z1, Z0};

  dim3 blk(256);
  for (int l = 0; l < 3; ++l) {
    const float* W   = P(4 + 6 * l);
    const float* a_s = P(5 + 6 * l);
    const float* a_d = P(6 + 6 * l);
    const float* b   = P(7 + 6 * l);
    const float* bng = P(8 + 6 * l);
    const float* bnb = P(9 + 6 * l);
    int Ci = Cin[l], Co = Cout[l];
    int total = NN * Co;

    // h = z @ W  (WMMA)
    gemm_bf16_wmma<<<dim3(Co / TN, NN / TM), blk, 0, stream>>>(zin[l], W, nullptr, HB, NN, Co, Ci, 0);
    // per-node attention scores
    node_scores<<<NN / 8, blk, 0, stream>>>(HB, a_s, a_d, ss, sd, Co);
    // init reductions + aggregation buffer
    fill_kernel<<<(NN + 255) / 256, blk, 0, stream>>>(mbuf, -INFINITY, NN);
    fill_kernel<<<(NN + 255) / 256, blk, 0, stream>>>(dbuf, 0.f, NN);
    fill_kernel<<<(total + 255) / 256, blk, 0, stream>>>(zout[l], 0.f, total);
    // edge softmax + aggregate
    edge_logits<<<(ET + 255) / 256, blk, 0, stream>>>(ei, ss, sd, ebuf, mbuf, EE, NN);
    edge_exp<<<(ET + 255) / 256, blk, 0, stream>>>(ei, ebuf, mbuf, exbuf, dbuf, EE, NN);
    edge_aggregate<<<(ET + 7) / 8, blk, 0, stream>>>(ei, HB, exbuf, dbuf, zout[l], EE, NN, Co);
    // bias + BN + ReLU
    bias_bn_relu<<<(total + 255) / 256, blk, 0, stream>>>(zout[l], b, bng, bnb, Co, total);
  }
  // final z (N x 128) is in Z0

  // ---- GAE head ----
  gemm_bf16_wmma<<<dim3(1, NN / TM), blk, 0, stream>>>(Z0, P(35), P(36), Z1, NN, 128, 128, 1); // r1
  gemm_bf16_wmma<<<dim3(1, NN / TM), blk, 0, stream>>>(Z1, P(37), P(38), HB, NN, 128, 128, 1); // r2
  adj_sigmoid_wmma<<<dim3(512 / TN, 512 / TM, GG), blk, 0, stream>>>(HB, (float*)d_out);

  // ---- classifier head (softmax over length-1 seq == 1, so enh == v; Wq/Wk dead) ----
  pool_mean<<<GG, 128, 0, stream>>>(Z0, pooled);
  head_gemm<<<(GG * 128 + 255) / 256, blk, 0, stream>>>(pooled, 128, nullptr, 0, P(24), nullptr, vbuf, GG, 128, 0);
  head_gemm<<<(GG * 128 + 255) / 256, blk, 0, stream>>>(pooled, 128, vbuf, 128, P(25), P(26), fused, GG, 128, 1);
  head_gemm<<<(GG * 64 + 255) / 256, blk, 0, stream>>>(fused, 128, gf, 7, P(27), P(28), h1, GG, 64, 1);
  head_gemm<<<(GG * 16 + 255) / 256, blk, 0, stream>>>(h1, 64, nullptr, 0, P(29), P(30), h2, GG, 16, 1);
  head_gemm<<<(GG * 8 + 255) / 256, blk, 0, stream>>>(h2, 16, nullptr, 0, P(31), P(32), h3, GG, 8, 1);
  float* logits = (float*)d_out + (size_t)GG * 512 * 512;
  head_gemm<<<(GG * 2 + 255) / 256, blk, 0, stream>>>(h3, 8, nullptr, 0, P(33), P(34), logits, GG, 2, 0);
}
